// idm_20813411516775
// MI455X (gfx1250) — matrix operand order repaired and checked by reading, not compiled, not run on hardware
//
#include <hip/hip_runtime.h>
#include <hip/hip_bf16.h>
#include <cstdint>
#include <cstddef>

typedef _Float16 half_t;
typedef __attribute__((ext_vector_type(16))) _Float16 v16h;
typedef __attribute__((ext_vector_type(8)))  _Float16 v8h;
typedef __attribute__((ext_vector_type(4)))  _Float16 v4h;
typedef __attribute__((ext_vector_type(8)))  float    v8f;
typedef __attribute__((ext_vector_type(4)))  float    v4f;

#define B_SZ   4
#define C_SZ   64
#define HF     64
#define WF     64
#define S_SZ   512
#define HID    256
#define OUT_H  256
#define OUT_W  256
#define Q_SZ   (OUT_H*OUT_W)
#define M_ALL  (B_SZ*HF*WF)          // 16384 rows of featT / G

// ---------------- WMMA helpers (CDNA5 16x16x32 f16 -> f32) ----------------
__device__ __forceinline__ v8f wmma16(v16h a, v16h b, v8f c) {
  return __builtin_amdgcn_wmma_f32_16x16x32_f16(false, a, false, b, (short)0, c, false, false);
}
// A fragment: lane holds M=(lane&15); K base = k0 + 16*(lane>=16); 16
// contiguous halves read as two 8-half runs at +0 and +8.
__device__ __forceinline__ v16h load_fragA(const half_t* p) {
  v8h lo = *(const v8h*)p;
  v8h hi = *(const v8h*)(p + 8);
  return __builtin_shufflevector(lo, hi, 0,1,2,3,4,5,6,7,8,9,10,11,12,13,14,15);
}
// B fragment from N-major storage [n][k]: lane n=(lane&15)+16*nt reads 16
// contiguous halves at k0 + 16*(lane>=16).
__device__ __forceinline__ v16h load_fragB(const half_t* p) {
  return *(const v16h*)p;
}

// ---------------- Phase A: modulation + demod ----------------
__global__ __launch_bounds__(64) void k_mod(const float* __restrict__ scale1,
                                            const float* __restrict__ mod_w,
                                            const float* __restrict__ mod_b,
                                            float* __restrict__ smod) {
  int b = blockIdx.x, c = threadIdx.x;
  float acc = 0.f;
  const float* sp = scale1 + b * S_SZ;
  const float* wp = mod_w + c * S_SZ;
  for (int k = 0; k < S_SZ; ++k) acc += sp[k] * wp[k];
  smod[b * C_SZ + c] = acc + mod_b[c];
}

__global__ __launch_bounds__(256) void k_demod(const float* __restrict__ conv_w,
                                               const float* __restrict__ smod,
                                               float* __restrict__ demod) {
  int t = blockIdx.x * 256 + threadIdx.x;      // 256 total = B*C
  int b = t >> 6, co = t & 63;
  float acc = 0.f;
  for (int ci = 0; ci < C_SZ; ++ci) {
    float sv = smod[b * C_SZ + ci];
    const float* wp = conv_w + (size_t)(co * C_SZ + ci) * 9;
#pragma unroll
    for (int i = 0; i < 9; ++i) { float w = wp[i] * sv; acc += w * w; }
  }
  demod[b * C_SZ + co] = rsqrtf(acc + 1e-8f);
}

// ---------------- param conversion: w1T f16 [256][64], w2T f16 [64][256],
// convwT f16 [9][cout=64][cin=64] ----------------
__global__ __launch_bounds__(256) void k_prep_w(const float* __restrict__ w1,
                                                const float* __restrict__ w2,
                                                const float* __restrict__ conv_w,
                                                half_t* __restrict__ w1T,
                                                half_t* __restrict__ w2T,
                                                half_t* __restrict__ convwT) {
  int idx = blockIdx.x * 256 + threadIdx.x;
  if (idx < 256 * 64) {                       // w1T[n][k] = w1[k][n]
    int n = idx >> 6, k = idx & 63;
    w1T[idx] = (half_t)w1[k * HID + n];
  } else if (idx < 2 * 256 * 64) {            // w2T[n][k] = w2[k][n]
    int j = idx - 256 * 64;
    int n = j >> 8, k = j & 255;
    w2T[j] = (half_t)w2[k * C_SZ + n];
  } else if (idx < 2 * 256 * 64 + 9 * 64 * 64) {
    int j = idx - 2 * 256 * 64;               // convwT[t][co][ci]
    int t = j >> 12, co = (j >> 6) & 63, ci = j & 63;
    convwT[j] = (half_t)conv_w[(size_t)(co * C_SZ + ci) * 9 + t];
  }
}

// ---------------- x*s -> f16 transposed [b][hw][c] ----------------
__global__ __launch_bounds__(256) void k_xs(const float* __restrict__ x,
                                            const float* __restrict__ smod,
                                            half_t* __restrict__ xsT) {
  int idx = blockIdx.x * 256 + threadIdx.x;   // B*4096*64 = 1048576
  int c = idx & 63, hw = (idx >> 6) & 4095, b = idx >> 18;
  float v = x[((size_t)(b * C_SZ + c)) * (HF * WF) + hw] * smod[b * C_SZ + c];
  xsT[idx] = (half_t)v;
}

// ---------------- conv 3x3 as implicit GEMM over 9 taps (WMMA) ----------------
// block: (b, row-pair). 8 waves; wave -> 16 pixels, 4 cout-tiles.
// writes yT f32 [b][hw][cout] = conv(x*s) * demod
__global__ __launch_bounds__(256) void k_conv(const half_t* __restrict__ xsT,
                                              const half_t* __restrict__ convwT,
                                              const float* __restrict__ demod,
                                              float* __restrict__ yT) {
  __shared__ half_t wsh[9 * 64 * 64];
  int t = threadIdx.x;
  for (int i = t; i < 9 * 64 * 64; i += 256) wsh[i] = convwT[i];
  __syncthreads();

  int blk = blockIdx.x;                        // 0..127
  int b = blk >> 5, rp = blk & 31;
  int wave = t >> 5, l = t & 31;
  int m = l & 15, hi = l >> 4;
  int row_local = wave >> 2, colbase = (wave & 3) * 16;
  int row = rp * 2 + row_local;
  int col = colbase + m;

  v8f acc[4];
#pragma unroll
  for (int nt = 0; nt < 4; ++nt) acc[nt] = {};

#pragma unroll
  for (int tap = 0; tap < 9; ++tap) {
    int dy = tap / 3 - 1, dx = tap % 3 - 1;
    int srow = row + dy, scol = col + dx;
    bool valid = (srow >= 0) & (srow < HF) & (scol >= 0) & (scol < WF);
    const half_t* abase =
        xsT + ((size_t)(b * HF * WF + srow * WF + scol)) * C_SZ;
#pragma unroll
    for (int ks = 0; ks < 2; ++ks) {
      v16h a;
      if (valid) {
        a = load_fragA(abase + ks * 32 + hi * 16);
      } else {
#pragma unroll
        for (int i = 0; i < 16; ++i) a[i] = (half_t)0.f;
      }
#pragma unroll
      for (int nt = 0; nt < 4; ++nt) {
        int n = nt * 16 + m;
        const half_t* pb = &wsh[(tap * 64 + n) * 64 + ks * 32 + hi * 16];
        acc[nt] = wmma16(a, load_fragB(pb), acc[nt]);
      }
    }
  }
  // D layout: VGPR r -> M = r + 8*hi, N = (l&15) + 16*nt
  float* yrow = yT + ((size_t)(b * HF * WF + row * WF + colbase)) * C_SZ;
#pragma unroll
  for (int nt = 0; nt < 4; ++nt) {
    int n = nt * 16 + (l & 15);
    float dm = demod[b * C_SZ + n];
#pragma unroll
    for (int r = 0; r < 8; ++r) {
      int M = hi * 8 + r;
      yrow[(size_t)M * C_SZ + n] = acc[nt][r] * dm;
    }
  }
}

// ---------------- instance-norm stats per (b,c) ----------------
__global__ __launch_bounds__(256) void k_stats(const float* __restrict__ yT,
                                               float* __restrict__ meanv,
                                               float* __restrict__ rstdv) {
  __shared__ float ssum[256], ssq[256];
  int b = blockIdx.x, t = threadIdx.x;
  int c = t & 63, part = t >> 6;
  float s = 0.f, q = 0.f;
  for (int hw = part; hw < HF * WF; hw += 4) {
    float v = yT[((size_t)(b * HF * WF + hw)) * C_SZ + c];
    s += v; q += v * v;
  }
  ssum[t] = s; ssq[t] = q;
  __syncthreads();
  if (t < 64) {
    float S = ssum[t] + ssum[t + 64] + ssum[t + 128] + ssum[t + 192];
    float Q = ssq[t] + ssq[t + 64] + ssq[t + 128] + ssq[t + 192];
    float mu = S * (1.f / (HF * WF));
    float var = Q * (1.f / (HF * WF)) - mu * mu;
    meanv[b * C_SZ + t] = mu;
    rstdv[b * C_SZ + t] = rsqrtf(var + 1e-5f);
  }
}

// ---------------- norm + affine + leaky_relu*sqrt2 -> featT f16 ----------------
__global__ __launch_bounds__(256) void k_feat(const float* __restrict__ yT,
                                              const float* __restrict__ meanv,
                                              const float* __restrict__ rstdv,
                                              const float* __restrict__ scale2,
                                              const float* __restrict__ shift,
                                              const float* __restrict__ act_b,
                                              half_t* __restrict__ featT) {
  int idx = blockIdx.x * 256 + threadIdx.x;   // B*4096*64
  int c = idx & 63, b = idx >> 18;
  float yv = yT[idx];
  float vn = (yv - meanv[b * C_SZ + c]) * rstdv[b * C_SZ + c];
  float v2 = vn * scale2[b * C_SZ + c] + shift[b * C_SZ + c] + act_b[c];
  float lr = v2 > 0.f ? v2 : 0.2f * v2;
  featT[idx] = (half_t)(lr * 1.41421356237f);
}

// ---------------- G = featT @ w1[:64] + b1   (f16 out, [M_ALL][256]) ----------------
__global__ __launch_bounds__(256) void k_gemm_g(const half_t* __restrict__ featT,
                                                const half_t* __restrict__ w1T,
                                                const float* __restrict__ b1,
                                                half_t* __restrict__ G) {
  int t = threadIdx.x;
  int wave = t >> 5, l = t & 31, m = l & 15, hi = l >> 4;
  int mrow = blockIdx.x * 128 + wave * 16;    // gridDim.x = 128
  int n0 = blockIdx.y * 64;                   // gridDim.y = 4
  v8f acc[4];
#pragma unroll
  for (int nt = 0; nt < 4; ++nt) acc[nt] = {};
#pragma unroll
  for (int ks = 0; ks < 2; ++ks) {
    const half_t* pa = featT + (size_t)(mrow + m) * C_SZ + ks * 32 + hi * 16;
    v16h a = load_fragA(pa);
#pragma unroll
    for (int nt = 0; nt < 4; ++nt) {
      int n = n0 + nt * 16 + m;
      const half_t* pb = w1T + (size_t)n * 64 + ks * 32 + hi * 16;
      acc[nt] = wmma16(a, load_fragB(pb), acc[nt]);
    }
  }
#pragma unroll
  for (int nt = 0; nt < 4; ++nt) {
    int n = n0 + nt * 16 + (l & 15);
    float bb = b1[n];
#pragma unroll
    for (int r = 0; r < 8; ++r) {
      int M = hi * 8 + r;
      G[(size_t)(mrow + M) * HID + n] = (half_t)(acc[nt][r] + bb);
    }
  }
}

// ---------------- main LIIF kernel: combine ensemble into H, then H @ w2 ----------------
__global__ __launch_bounds__(256) void k_main(const half_t* __restrict__ G,
                                              const float* __restrict__ w1, // rows 64,65 = u,v
                                              const half_t* __restrict__ w2T,
                                              const float* __restrict__ b2,
                                              float* __restrict__ out) {
  __shared__ half_t Hs[128][HID + 8];
  __shared__ half_t w2s[64 * HID];
  int t = threadIdx.x;
  int blk = blockIdx.x;                        // 2048 blocks
  int b = blk >> 9;
  int qbase = (blk & 511) << 7;                // 128 queries per tile

  for (int i = t; i < 64 * HID; i += 256) w2s[i] = w2T[i];

  // -------- build combined hidden vector H for 128 queries --------
  int ql = t >> 1;
  int khalf = (t & 1) << 7;                    // 0 or 128
  int q = qbase + ql;
  int oy = q >> 8, ox = q & 255;
  float cy0 = -1.f + (2.f * oy + 1.f) * (1.f / OUT_H);
  float cx0 = -1.f + (2.f * ox + 1.f) * (1.f / OUT_W);
  const float rx = 1.f / HF, ry = 1.f / WF;

  int cell[4]; float area[4], rely[4], relx[4];
#pragma unroll
  for (int s = 0; s < 4; ++s) {
    float vx = (s >> 1) ? 1.f : -1.f;
    float vy = (s & 1) ? 1.f : -1.f;
    float cy = fminf(fmaxf(cy0 + vx * rx + 1e-6f, -1.f + 1e-6f), 1.f - 1e-6f);
    float cx = fminf(fmaxf(cx0 + vy * ry + 1e-6f, -1.f + 1e-6f), 1.f - 1e-6f);
    int iy = min(max((int)floorf((cy + 1.f) * 0.5f * HF), 0), HF - 1);
    int ix = min(max((int)floorf((cx + 1.f) * 0.5f * WF), 0), WF - 1);
    cell[s] = iy * WF + ix;
    float qy = -1.f + (2.f * iy + 1.f) * (1.f / HF);
    float qx = -1.f + (2.f * ix + 1.f) * (1.f / WF);
    rely[s] = (cy0 - qy) * HF;
    relx[s] = (cx0 - qx) * WF;
    area[s] = fabsf(rely[s] * relx[s]) + 1e-9f;
  }
  float inv_tot = 1.f / (area[0] + area[1] + area[2] + area[3]);
  float w4[4], wry = 0.f, wrx = 0.f;
#pragma unroll
  for (int s = 0; s < 4; ++s) {
    w4[s] = area[3 - s] * inv_tot;             // diagonal-swapped ensemble weight
    wry += w4[s] * rely[s];
    wrx += w4[s] * relx[s];
  }
  const half_t* G0 = G + (size_t)(b * HF * WF + cell[0]) * HID + khalf;
  const half_t* G1 = G + (size_t)(b * HF * WF + cell[1]) * HID + khalf;
  const half_t* G2 = G + (size_t)(b * HF * WF + cell[2]) * HID + khalf;
  const half_t* G3 = G + (size_t)(b * HF * WF + cell[3]) * HID + khalf;
  const float* up = w1 + 64 * HID + khalf;
  const float* vp = w1 + 65 * HID + khalf;
  for (int k = 0; k < 128; k += 4) {
    v4h g0 = *(const v4h*)(G0 + k);
    v4h g1 = *(const v4h*)(G1 + k);
    v4h g2 = *(const v4h*)(G2 + k);
    v4h g3 = *(const v4h*)(G3 + k);
    v4f uu = *(const v4f*)(up + k);
    v4f vv = *(const v4f*)(vp + k);
    v4h hout;
#pragma unroll
    for (int e = 0; e < 4; ++e) {
      float h = w4[0] * (float)g0[e] + w4[1] * (float)g1[e] +
                w4[2] * (float)g2[e] + w4[3] * (float)g3[e] +
                wry * uu[e] + wrx * vv[e];
      hout[e] = (half_t)h;
    }
    *(v4h*)&Hs[ql][khalf + k] = hout;
  }
  __syncthreads();

  // -------- GEMM: [128 x 256] f16 @ [256 x 64] f16 -> f32 --------
  int wave = t >> 5, l = t & 31, m = l & 15, hi = l >> 4;
  v8f acc[4];
#pragma unroll
  for (int nt = 0; nt < 4; ++nt) acc[nt] = {};
#pragma unroll
  for (int kk = 0; kk < 8; ++kk) {
    int k0 = kk * 32;
    v16h a = load_fragA(&Hs[wave * 16 + m][k0 + hi * 16]);
#pragma unroll
    for (int nt = 0; nt < 4; ++nt) {
      const half_t* pb = &w2s[(nt * 16 + m) * HID + k0 + hi * 16];
      acc[nt] = wmma16(a, load_fragB(pb), acc[nt]);
    }
  }
  float* outb = out + ((size_t)b * Q_SZ + qbase) * C_SZ;
#pragma unroll
  for (int nt = 0; nt < 4; ++nt) {
    int n = nt * 16 + (l & 15);
    float bb = b2[n];
#pragma unroll
    for (int r = 0; r < 8; ++r) {
      int qrow = wave * 16 + hi * 8 + r;
      outb[(size_t)qrow * C_SZ + n] = acc[nt][r] + bb;
    }
  }
}

// ---------------- launcher ----------------
extern "C" void kernel_launch(void* const* d_in, const int* in_sizes, int n_in,
                              void* d_out, int out_size, void* d_ws, size_t ws_size,
                              hipStream_t stream) {
  const float* x      = (const float*)d_in[0];
  const float* scale1 = (const float*)d_in[1];
  const float* scale2 = (const float*)d_in[2];
  const float* shiftp = (const float*)d_in[3];
  const float* mod_w  = (const float*)d_in[4];
  const float* mod_b  = (const float*)d_in[5];
  const float* conv_w = (const float*)d_in[6];
  const float* act_b  = (const float*)d_in[7];
  const float* w1     = (const float*)d_in[8];
  const float* b1     = (const float*)d_in[9];
  const float* w2     = (const float*)d_in[10];
  const float* b2     = (const float*)d_in[11];
  float* out = (float*)d_out;

  char* ws = (char*)d_ws;
  float*  smod   = (float*)(ws + 0);                 //   1 KB
  float*  demod  = (float*)(ws + 1024);              //   1 KB
  float*  meanv  = (float*)(ws + 2048);              //   1 KB
  float*  rstdv  = (float*)(ws + 3072);              //   1 KB
  half_t* w1T    = (half_t*)(ws + 4096);             //  32 KB
  half_t* w2T    = (half_t*)(ws + 36864);            //  32 KB
  half_t* convwT = (half_t*)(ws + 69632);            //  72 KB
  half_t* xsT    = (half_t*)(ws + 143360);           //   2 MB
  float*  yT     = (float*)(ws + 2240512);           //   4 MB
  half_t* featT  = (half_t*)(ws + 6434816);          //   2 MB
  half_t* G      = (half_t*)(ws + 8531968);          //   8 MB  (total ~16.2 MB)

  k_mod   <<<B_SZ, 64, 0, stream>>>(scale1, mod_w, mod_b, smod);
  k_demod <<<1, 256, 0, stream>>>(conv_w, smod, demod);
  k_prep_w<<<(2 * 256 * 64 + 9 * 64 * 64 + 255) / 256, 256, 0, stream>>>(
      w1, w2, conv_w, w1T, w2T, convwT);
  k_xs    <<<(B_SZ * HF * WF * C_SZ) / 256, 256, 0, stream>>>(x, smod, xsT);
  k_conv  <<<B_SZ * (HF / 2), 256, 0, stream>>>(xsT, convwT, demod, yT);
  k_stats <<<B_SZ, 256, 0, stream>>>(yT, meanv, rstdv);
  k_feat  <<<(B_SZ * HF * WF * C_SZ) / 256, 256, 0, stream>>>(
      yT, meanv, rstdv, scale2, shiftp, act_b, featT);
  k_gemm_g<<<dim3(M_ALL / 128, HID / 64), 256, 0, stream>>>(featT, w1T, b1, G);
  k_main  <<<B_SZ * (Q_SZ / 128), 256, 0, stream>>>(G, w1, w2T, b2, out);
}
